// CGCNN_53970559042215
// MI455X (gfx1250) — matrix operand order repaired and checked by reading, not compiled, not run on hardware
//
#include <hip/hip_runtime.h>
#include <stdint.h>

// ---------------------------------------------------------------------------
// CGCNN pipeline for MI455X (gfx1250, wave32, WMMA).
// Heavy path: per-edge gated GEMM  z[E,178] @ {Wf,Ws}[178,64]  via
// v_wmma_f32_16x16x32_f16 (f16 inputs, f32 accum), z staged in LDS,
// weights pre-converted to f16 transposed [n][k] (B-fragment layout),
// segment-sum via global_atomic_add_f32 (aggr fits in 192MB L2).
// Round 2: batched fragment preloads (one s_wait_dscnt per k-step instead of
// per-wmma) and packed b32 LDS staging stores.
// ---------------------------------------------------------------------------

typedef __attribute__((ext_vector_type(16))) _Float16 v16h;
typedef __attribute__((ext_vector_type(8)))  _Float16 v8h;
typedef __attribute__((ext_vector_type(2)))  _Float16 v2h;
typedef __attribute__((ext_vector_type(8)))  float    v8f;

#define N_NODES 30000
#define N_EDGES 480000
#define NFEAT   92
#define EFEAT   50
#define D1      64
#define GB      256
#define ZK_E    192   // 2*D1+EF = 178 padded to 6 k-steps of 32
#define ZK_0    96    // NF = 92 padded to 3 k-steps of 32
#define NT_E    (N_EDGES / 16)   // 30000 edge tiles
#define NT_0    (N_NODES / 16)   // 1875 node tiles

__device__ __forceinline__ v8f wmma_f16(v16h a, v16h b, v8f c) {
  // 8 args: (neg_a, A, neg_b, B, c_mod, C, reuse_a, reuse_b)
  return __builtin_amdgcn_wmma_f32_16x16x32_f16(false, a, false, b, (short)0, c,
                                                false, false);
}

// A-fragment 16x32 f16 from a row-major LDS tile.
// lane&15 = row, lane>>4 selects which contiguous 16-K half.
__device__ __forceinline__ v16h load_a_frag(const _Float16* zrow, int kb, int hi) {
  v8h c0 = *(const v8h*)(zrow + kb + hi * 16);
  v8h c1 = *(const v8h*)(zrow + kb + hi * 16 + 8);
  v16h a;
#pragma unroll
  for (int i = 0; i < 8; ++i) { a[i] = c0[i]; a[i + 8] = c1[i]; }
  return a;
}

// ------------------------- weight prep (f32 -> f16, transposed, padded) ----
// Layout in wbuf (halves): [W0t 64x96][Wf0 64x192][Ws0][Wf1][Ws1][Wf2][Ws2]
__global__ void k_prep(const float* __restrict__ lin0_w,
                       const float* __restrict__ cwf,
                       const float* __restrict__ cws,
                       _Float16* __restrict__ wbuf) {
  const int S0 = 64 * ZK_0;      // 6144
  const int SM = 64 * ZK_E;      // 12288
  const int TOT = S0 + 6 * SM;   // 79872
  int i = blockIdx.x * blockDim.x + threadIdx.x;
  if (i >= TOT) return;
  if (i < S0) {
    int n = i / ZK_0, k = i % ZK_0;
    wbuf[i] = (k < NFEAT) ? (_Float16)lin0_w[k * D1 + n] : (_Float16)0.0f;
  } else {
    int j = i - S0;
    int mat = j / SM;            // 0..5 : (layer<<1)|isWs
    int r   = j % SM;
    int n = r / ZK_E, k = r % ZK_E;
    int l = mat >> 1;
    const float* src = (mat & 1) ? cws : cwf;
    wbuf[i] = (k < 178) ? (_Float16)src[(l * 178 + k) * D1 + n] : (_Float16)0.0f;
  }
}

// ------------------------------ misc small kernels -------------------------
__global__ void k_zero(float* __restrict__ p, int n) {
  int i = blockIdx.x * blockDim.x + threadIdx.x;
  if (i < n) p[i] = 0.0f;
}

__global__ void k_count(const int* __restrict__ ei, float* __restrict__ cnt) {
  int e = blockIdx.x * blockDim.x + threadIdx.x;
  if (e < N_EDGES) atomicAdd(&cnt[ei[N_EDGES + e]], 1.0f);  // dst = edge_index[1]
}

// ------------------------------ lin0 : h = relu(x @ W0 + b0) ---------------
struct SharedL0 {
  alignas(32) _Float16 w0[64 * ZK_0];    // 12 KB, transposed [n][k]
  alignas(32) _Float16 z[4][16 * ZK_0];  // 4 waves x 3 KB
};

__global__ void __launch_bounds__(128)
k_lin0(const float* __restrict__ x, const _Float16* __restrict__ w0t,
       const float* __restrict__ b0, float* __restrict__ h) {
  __shared__ SharedL0 sh;
  const int tid = threadIdx.x;
  for (int i = tid; i < 64 * ZK_0 / 8; i += blockDim.x)
    ((uint4*)sh.w0)[i] = ((const uint4*)w0t)[i];
  __syncthreads();

  const int wave = tid >> 5, lane = tid & 31;
  const int r16 = lane & 15, hi = lane >> 4;

  for (int tile = blockIdx.x * 4 + wave; tile < NT_0; tile += gridDim.x * 4) {
    _Float16* zt = sh.z[wave];
    // stage 16 x 96 f16 tile of x (packed pair stores; x rows 8B aligned)
    for (int r = 0; r < 16; ++r) {
      const float* xr = x + (size_t)(tile * 16 + r) * NFEAT;
      _Float16* zr = zt + r * ZK_0;
      for (int p = lane; p < ZK_0 / 2; p += 32) {   // 48 pairs, 16 exact/lane? 48/32 -> 2 iters ragged but uniform <48 check
        v2h pr;
        if (p < NFEAT / 2) {                        // cols 0..91, NFEAT even
          float2 v = *(const float2*)(xr + 2 * p);
          pr[0] = (_Float16)v.x; pr[1] = (_Float16)v.y;
        } else {
          pr[0] = (_Float16)0.0f; pr[1] = (_Float16)0.0f;
        }
        *(v2h*)(zr + 2 * p) = pr;
      }
    }
    v8f acc[4] = {};
#pragma unroll
    for (int ks = 0; ks < 3; ++ks) {
      const int kb = ks * 32;
      v16h a = load_a_frag(zt + r16 * ZK_0, kb, hi);
      v16h b[4];
#pragma unroll
      for (int nt = 0; nt < 4; ++nt)
        b[nt] = *(const v16h*)&sh.w0[(nt * 16 + r16) * ZK_0 + kb + hi * 16];
#pragma unroll
      for (int nt = 0; nt < 4; ++nt)
        acc[nt] = wmma_f16(a, b[nt], acc[nt]);
    }
#pragma unroll
    for (int nt = 0; nt < 4; ++nt) {
      int c = nt * 16 + r16;
      float bb = b0[c];
#pragma unroll
      for (int v = 0; v < 8; ++v) {
        int node = tile * 16 + v + 8 * hi;
        float t = acc[nt][v] + bb;
        h[(size_t)node * D1 + c] = t > 0.0f ? t : 0.0f;
      }
    }
  }
}

// ----------------- edge messages: m = sigmoid(zWf+bf)*softplus(zWs+bs) -----
#define EWAVES 2
struct SharedE {
  alignas(32) _Float16 wf[64 * ZK_E];        // 24 KB
  alignas(32) _Float16 wsm[64 * ZK_E];       // 24 KB
  alignas(32) _Float16 z[EWAVES][16 * ZK_E]; // 2 x 6 KB
  int dstIdx[EWAVES][16];
};                                           // ~60.3 KB

__global__ void __launch_bounds__(32 * EWAVES)
k_edge(const float* __restrict__ h, const float* __restrict__ edge_attr,
       const int* __restrict__ ei, const _Float16* __restrict__ wfT,
       const _Float16* __restrict__ wsT, const float* __restrict__ bfp,
       const float* __restrict__ bsp, float* __restrict__ aggr) {
  __shared__ SharedE sh;
  const int tid = threadIdx.x;
  for (int i = tid; i < 64 * ZK_E / 8; i += blockDim.x) {
    ((uint4*)sh.wf)[i]  = ((const uint4*)wfT)[i];
    ((uint4*)sh.wsm)[i] = ((const uint4*)wsT)[i];
  }
  __syncthreads();

  const int wave = tid >> 5, lane = tid & 31;
  const int r16 = lane & 15, hi = lane >> 4;

  for (int tile = blockIdx.x * EWAVES + wave; tile < NT_E;
       tile += gridDim.x * EWAVES) {
    _Float16* zt = sh.z[wave];
    // stage z = [h[dst] | h[src] | edge_attr | 0-pad] as 16x192 f16,
    // packed pair stores (all segment boundaries even). 96 pairs/row,
    // 3 pairs/lane/row.
    for (int r = 0; r < 16; ++r) {
      int e  = tile * 16 + r;
      int sN = ei[e];             // src = edge_index[0]
      int dN = ei[N_EDGES + e];   // dst = edge_index[1]
      if (lane == 0) sh.dstIdx[wave][r] = dN;
      const float* hd = h + (size_t)dN * D1;
      const float* hs = h + (size_t)sN * D1;
      const float* ea = edge_attr + (size_t)e * EFEAT;
      _Float16* zr = zt + r * ZK_E;
#pragma unroll
      for (int pp = 0; pp < 3; ++pp) {
        int p = lane + 32 * pp;   // pair index 0..95
        v2h pr;
        if (p < 32) {             // h[dst] cols 0..63 (8B aligned rows)
          float2 v = *(const float2*)(hd + 2 * p);
          pr[0] = (_Float16)v.x; pr[1] = (_Float16)v.y;
        } else if (p < 64) {      // h[src] cols 64..127
          float2 v = *(const float2*)(hs + 2 * (p - 32));
          pr[0] = (_Float16)v.x; pr[1] = (_Float16)v.y;
        } else if (p < 89) {      // edge_attr cols 128..177 (8B aligned rows)
          int c0 = 2 * (p - 64);
          pr[0] = (_Float16)ea[c0]; pr[1] = (_Float16)ea[c0 + 1];
        } else {                  // pad cols 178..191
          pr[0] = (_Float16)0.0f; pr[1] = (_Float16)0.0f;
        }
        *(v2h*)(zr + 2 * p) = pr;
      }
    }
    v8f accF[4] = {}, accS[4] = {};
#pragma unroll
    for (int ks = 0; ks < 6; ++ks) {
      const int kb = ks * 32;
      // Preload A + all 8 B fragments, THEN issue 8 wmmas back-to-back:
      // lets the scheduler cover the k-step with a single s_wait_dscnt.
      v16h a = load_a_frag(zt + r16 * ZK_E, kb, hi);
      v16h bF[4], bS[4];
#pragma unroll
      for (int nt = 0; nt < 4; ++nt) {
        const int wo = (nt * 16 + r16) * ZK_E + kb + hi * 16;
        bF[nt] = *(const v16h*)&sh.wf[wo];
        bS[nt] = *(const v16h*)&sh.wsm[wo];
      }
#pragma unroll
      for (int nt = 0; nt < 4; ++nt) accF[nt] = wmma_f16(a, bF[nt], accF[nt]);
#pragma unroll
      for (int nt = 0; nt < 4; ++nt) accS[nt] = wmma_f16(a, bS[nt], accS[nt]);
    }
    // epilogue: gate, then scatter-add to aggr[dst]
#pragma unroll
    for (int nt = 0; nt < 4; ++nt) {
      int c = nt * 16 + r16;
      float bf = bfp[c], bs = bsp[c];
#pragma unroll
      for (int v = 0; v < 8; ++v) {
        int m = v + 8 * hi;
        float f  = accF[nt][v] + bf;
        float s  = accS[nt][v] + bs;
        float sg = 1.0f / (1.0f + __expf(-f));
        float sp = (s > 20.0f) ? s : log1pf(__expf(s));
        int dN = sh.dstIdx[wave][m];
        atomicAdd(&aggr[(size_t)dN * D1 + c], sg * sp);
      }
    }
  }
}

// --------- BN stats: aggr /= cnt (in place), channel sum & sumsq -----------
__global__ void __launch_bounds__(256)
k_stats(const float* __restrict__ cnt, float* __restrict__ aggr,
        float* __restrict__ csum, float* __restrict__ csum2) {
  __shared__ float rs[256], rs2[256];
  const int tid = threadIdx.x;
  const int c = tid & 63, rg = tid >> 6;
  float s = 0.0f, s2 = 0.0f;
  for (int n = blockIdx.x * 4 + rg; n < N_NODES; n += gridDim.x * 4) {
    float inv = 1.0f / fmaxf(cnt[n], 1.0f);
    float a = aggr[(size_t)n * D1 + c] * inv;
    aggr[(size_t)n * D1 + c] = a;
    s += a; s2 += a * a;
  }
  rs[tid] = s; rs2[tid] = s2;
  __syncthreads();
  if (tid < 64) {
    atomicAdd(&csum[tid],  rs[tid] + rs[tid + 64] + rs[tid + 128] + rs[tid + 192]);
    atomicAdd(&csum2[tid], rs2[tid] + rs2[tid + 64] + rs2[tid + 128] + rs2[tid + 192]);
  }
}

// ---------------- h = relu(h + gamma*(a-mu)*rsqrt(var+eps) + beta) ---------
__global__ void k_update(float* __restrict__ h, const float* __restrict__ aggr,
                         const float* __restrict__ csum, const float* __restrict__ csum2,
                         const float* __restrict__ gamma, const float* __restrict__ beta) {
  int i = blockIdx.x * blockDim.x + threadIdx.x;
  if (i >= N_NODES * D1) return;
  int c = i & 63;
  const float invN = 1.0f / (float)N_NODES;
  float mu  = csum[c] * invN;
  float var = csum2[c] * invN - mu * mu;
  float sc  = gamma[c] * rsqrtf(var + 1e-5f);
  float t = h[i] + (aggr[i] - mu) * sc + beta[c];
  h[i] = t > 0.0f ? t : 0.0f;
}

// --------------------------- global mean pool ------------------------------
__global__ void k_pool(const float* __restrict__ h, const int* __restrict__ batch,
                       float* __restrict__ pooled, float* __restrict__ gcnt) {
  int i = blockIdx.x * blockDim.x + threadIdx.x;
  if (i >= N_NODES * D1) return;
  int n = i >> 6, c = i & 63;
  int g = batch[n];
  atomicAdd(&pooled[g * D1 + c], h[i]);
  if (c == 0) atomicAdd(&gcnt[g], 1.0f);
}

__global__ void k_pool_norm(float* __restrict__ pooled, const float* __restrict__ gcnt) {
  int i = blockIdx.x * blockDim.x + threadIdx.x;
  if (i >= GB * D1) return;
  pooled[i] /= fmaxf(gcnt[i >> 6], 1.0f);
}

// --------------------------- MLP head (tiny) -------------------------------
__global__ void k_dense(const float* __restrict__ in, const float* __restrict__ W,
                        const float* __restrict__ b, float* __restrict__ out,
                        int K, int J, int doRelu) {
  int i = blockIdx.x * blockDim.x + threadIdx.x;
  if (i >= GB * J) return;
  int g = i / J, j = i % J;
  float s = b[j];
  for (int k = 0; k < K; ++k) s += in[g * K + k] * W[k * J + j];
  out[i] = doRelu ? fmaxf(s, 0.0f) : s;
}

// ---------------------------------------------------------------------------
extern "C" void kernel_launch(void* const* d_in, const int* in_sizes, int n_in,
                              void* d_out, int out_size, void* d_ws, size_t ws_size,
                              hipStream_t stream) {
  const float* x         = (const float*)d_in[0];
  const float* edge_attr = (const float*)d_in[1];
  const float* lin0_w    = (const float*)d_in[2];
  const float* lin0_b    = (const float*)d_in[3];
  const float* conv_wf   = (const float*)d_in[4];
  const float* conv_bf   = (const float*)d_in[5];
  const float* conv_ws   = (const float*)d_in[6];
  const float* conv_bs   = (const float*)d_in[7];
  const float* bn_gamma  = (const float*)d_in[8];
  const float* bn_beta   = (const float*)d_in[9];
  const float* lin1_w    = (const float*)d_in[10];
  const float* lin1_b    = (const float*)d_in[11];
  const float* fc_w      = (const float*)d_in[12];
  const float* fc_b      = (const float*)d_in[13];
  const float* lin2_w    = (const float*)d_in[14];
  const float* lin2_b    = (const float*)d_in[15];
  const int*   ei        = (const int*)d_in[16];
  const int*   batch     = (const int*)d_in[17];

  // ---- workspace arena (all offsets 256B aligned) ----
  char* w = (char*)d_ws;
  float*     hbuf   = (float*)(w + 0);                 // 7,680,000
  float*     aggr   = (float*)(w + 7680000);           // 7,680,000
  float*     cnt    = (float*)(w + 15360000);          //   120,000
  float*     csum   = (float*)(w + 15480064);          //       256
  float*     csum2  = (float*)(w + 15480320);          //       256
  float*     pooled = (float*)(w + 15480576);          //    65,536
  float*     gcnt   = (float*)(w + 15546112);          //     1,024
  float*     buf1   = (float*)(w + 15547136);          //    65,536
  float*     buf2   = (float*)(w + 15612672);          //    65,536
  _Float16*  wbuf   = (_Float16*)(w + 15678208);       //   159,744

  const _Float16* w0t = wbuf;                                   // 64x96
  auto wfT = [&](int l) { return wbuf + 6144 + (size_t)(2 * l) * 12288; };
  auto wsT = [&](int l) { return wbuf + 6144 + (size_t)(2 * l + 1) * 12288; };

  // 1) weights -> f16 transposed/padded
  k_prep<<<(79872 + 255) / 256, 256, 0, stream>>>(lin0_w, conv_wf, conv_ws, wbuf);

  // 2) per-dst edge counts
  k_zero<<<(N_NODES + 255) / 256, 256, 0, stream>>>(cnt, N_NODES);
  k_count<<<(N_EDGES + 255) / 256, 256, 0, stream>>>(ei, cnt);

  // 3) h = relu(x @ lin0_w + b)
  k_lin0<<<256, 128, 0, stream>>>(x, w0t, lin0_b, hbuf);

  // 4) three CGConv + BN + residual-ReLU layers
  for (int l = 0; l < 3; ++l) {
    k_zero<<<(N_NODES * D1 + 255) / 256, 256, 0, stream>>>(aggr, N_NODES * D1);
    k_zero<<<1, 128, 0, stream>>>(csum, 128);  // csum + csum2 contiguous
    k_edge<<<512, 32 * EWAVES, 0, stream>>>(hbuf, edge_attr, ei, wfT(l), wsT(l),
                                            conv_bf + l * D1, conv_bs + l * D1, aggr);
    k_stats<<<128, 256, 0, stream>>>(cnt, aggr, csum, csum2);
    k_update<<<(N_NODES * D1 + 255) / 256, 256, 0, stream>>>(
        hbuf, aggr, csum, csum2, bn_gamma + l * D1, bn_beta + l * D1);
  }

  // 5) global mean pool
  k_zero<<<(GB * D1 + GB + 255) / 256, 256, 0, stream>>>(pooled, GB * D1 + GB);
  k_pool<<<(N_NODES * D1 + 255) / 256, 256, 0, stream>>>(hbuf, batch, pooled, gcnt);
  k_pool_norm<<<(GB * D1 + 255) / 256, 256, 0, stream>>>(pooled, gcnt);

  // 6) MLP head
  k_dense<<<(GB * D1 + 255) / 256, 256, 0, stream>>>(pooled, lin1_w, lin1_b, buf1, 64, 64, 1);
  k_dense<<<(GB * D1 + 255) / 256, 256, 0, stream>>>(buf1, fc_w,            fc_b,      buf2, 64, 64, 1);
  k_dense<<<(GB * D1 + 255) / 256, 256, 0, stream>>>(buf2, fc_w + 64 * 64,  fc_b + 64, buf1, 64, 64, 1);
  k_dense<<<(GB + 255) / 256, 256, 0, stream>>>(buf1, lin2_w, lin2_b, (float*)d_out, 64, 1, 0);
}
